// GCNFeatureExtractor_85684597555829
// MI455X (gfx1250) — matrix operand order repaired and checked by reading, not compiled, not run on hardware
//
#include <hip/hip_runtime.h>

#define NNODES 50000
#define FEAT   128

typedef __attribute__((ext_vector_type(16))) __bf16 v16bf;
typedef __attribute__((ext_vector_type(8)))  __bf16 v8bf;
typedef __attribute__((ext_vector_type(8)))  float  v8f;

__device__ __forceinline__ unsigned short f2bf(float f) {
    unsigned int u = __float_as_uint(f);
    // round-to-nearest-even
    unsigned int r = (u + 0x7FFFu + ((u >> 16) & 1u)) >> 16;
    return (unsigned short)r;
}

// native fp32 atomic add (global_atomic_add_f32, no return -> STOREcnt path)
__device__ __forceinline__ void atomic_add_f32(float* p, float v) {
    __hip_atomic_fetch_add(p, v, __ATOMIC_RELAXED, __HIP_MEMORY_SCOPE_AGENT);
}

// ---------------- degree / normalization ----------------
__global__ void deg_init_kernel(float* deg, int n) {
    int i = blockIdx.x * blockDim.x + threadIdx.x;
    if (i < n) deg[i] = 1.0f;   // self-loop contribution
}

__global__ void deg_accum_kernel(const long long* dst, float* deg, int E) {
    int e = blockIdx.x * blockDim.x + threadIdx.x;
    if (e < E) atomic_add_f32(&deg[(int)dst[e]], 1.0f);
}

__global__ void deg_rsqrt_kernel(float* deg, int n) {
    int i = blockIdx.x * blockDim.x + threadIdx.x;
    if (i < n) deg[i] = rsqrtf(deg[i]);   // deg >= 1 always (self loop)
}

// ---------------- fp32 -> bf16 conversion ----------------
__global__ void cvt_f32_bf16_kernel(const float* in, unsigned short* out, int n) {
    int i = blockIdx.x * blockDim.x + threadIdx.x;
    if (i < n) out[i] = f2bf(in[i]);
}

// ---------------- pack W (128x128, row-major K x N) into per-lane WMMA B fragments
// Storage index tid = ((t*4 + kt)*32 + lane)*16 + j
//   t  = output col tile (0..7), kt = K tile (0..3), lane = wave lane, j = 0..15
//   value = W[(kt*32 + (lane>>4)*8 + (j&7) + (j>>3)*16) * 128 + t*16 + (lane&15)]
__global__ void pack_w_kernel(const float* W, unsigned short* Wp) {
    int tid  = blockIdx.x * blockDim.x + threadIdx.x;   // 0..16383
    int j    = tid & 15;
    int lane = (tid >> 4) & 31;
    int kt   = (tid >> 9) & 3;
    int t    = tid >> 11;
    int k = kt * 32 + (lane >> 4) * 8 + (j & 7) + (j >> 3) * 16;
    int n = t * 16 + (lane & 15);
    Wp[tid] = f2bf(W[k * FEAT + n]);
}

// ---------------- bf16 WMMA GEMM: out[M x 128] = A[M x 128] * W[128 x 128]
// grid.x = M/16, block = 256 (8 waves); wave w computes the 16x16 tile at cols w*16.
// No divergence: EXEC is all ones as required by WMMA.
__global__ void __launch_bounds__(256) gemm_bf16_wmma_kernel(
        const unsigned short* __restrict__ Abf,
        const unsigned short* __restrict__ Wp,
        float* __restrict__ out) {
    const int wave  = threadIdx.x >> 5;        // col tile 0..7
    const int lane  = threadIdx.x & 31;
    const int row0  = blockIdx.x * 16;
    const int m     = row0 + (lane & 15);
    const int khalf = lane >> 4;               // 0: K 0-7/16-23, 1: K 8-15/24-31

    const __bf16* arow = (const __bf16*)(Abf + (size_t)m * FEAT);
    const __bf16* bp   = (const __bf16*)(Wp + ((size_t)wave * 4 * 32 + lane) * 16);

    v8f c = {};
#pragma unroll
    for (int kt = 0; kt < 4; ++kt) {
        const int kbase = kt * 32 + khalf * 8;
        v8bf lo = *(const v8bf*)(arow + kbase);        // K = kbase .. kbase+7
        v8bf hi = *(const v8bf*)(arow + kbase + 16);   // K = kbase+16 .. kbase+23
        v16bf a;
#pragma unroll
        for (int j = 0; j < 8; ++j) { a[j] = lo[j]; a[8 + j] = hi[j]; }
        v16bf b = *(const v16bf*)(bp + (size_t)kt * 32 * 16);
        c = __builtin_amdgcn_wmma_f32_16x16x32_bf16(
                false, a, false, b, (short)0, c, false, false);
    }

    // C/D layout: lane col = wave*16 + (lane&15); VGPR r holds row r + 8*khalf
    float* orow = out + (size_t)row0 * FEAT + wave * 16 + (lane & 15);
#pragma unroll
    for (int r = 0; r < 8; ++r)
        orow[(size_t)(r + khalf * 8) * FEAT] = c[r];
}

// ---------------- self-loop init: agg[i,:] = h[i,:] * dinv[i]^2 ----------------
__global__ void self_init_kernel(const float* h, const float* dinv, float* agg, int total) {
    int t = blockIdx.x * blockDim.x + threadIdx.x;
    if (t >= total) return;
    int i = t >> 7;
    float di = dinv[i];
    agg[t] = h[t] * di * di;
}

// ---------------- edge scatter: one wave per edge, float4 per lane ----------------
__global__ void edge_scatter_kernel(const long long* __restrict__ src,
                                    const long long* __restrict__ dst,
                                    const float* __restrict__ dinv,
                                    const float* __restrict__ h,
                                    float* __restrict__ agg, int E) {
    int gid  = blockIdx.x * blockDim.x + threadIdx.x;
    int e    = gid >> 5;
    int lane = gid & 31;
    if (e >= E) return;
    int s = (int)src[e];
    int d = (int)dst[e];
    float nrm = dinv[s] * dinv[d];
    const float4* hp = (const float4*)(h + (size_t)s * FEAT);
    float*        op = agg + (size_t)d * FEAT + lane * 4;
    float4 v = hp[lane];
    atomic_add_f32(op + 0, v.x * nrm);
    atomic_add_f32(op + 1, v.y * nrm);
    atomic_add_f32(op + 2, v.z * nrm);
    atomic_add_f32(op + 3, v.w * nrm);
}

// ---------------- bias + relu, emit bf16 for next GEMM ----------------
__global__ void bias_relu_bf16_kernel(const float* agg, const float* b,
                                      unsigned short* out_bf, int total) {
    int t = blockIdx.x * blockDim.x + threadIdx.x;
    if (t >= total) return;
    float v = agg[t] + b[t & (FEAT - 1)];
    out_bf[t] = f2bf(v > 0.0f ? v : 0.0f);
}

// ---------------- final bias add (in place on d_out) ----------------
__global__ void bias_add_kernel(float* agg, const float* b, int total) {
    int t = blockIdx.x * blockDim.x + threadIdx.x;
    if (t >= total) return;
    agg[t] += b[t & (FEAT - 1)];
}

extern "C" void kernel_launch(void* const* d_in, const int* in_sizes, int n_in,
                              void* d_out, int out_size, void* d_ws, size_t ws_size,
                              hipStream_t stream) {
    const float*     x  = (const float*)d_in[0];
    const long long* ei = (const long long*)d_in[1];
    const float*     W1 = (const float*)d_in[2];
    const float*     b1 = (const float*)d_in[3];
    const float*     W2 = (const float*)d_in[4];
    const float*     b2 = (const float*)d_in[5];
    float* out = (float*)d_out;

    const int E = in_sizes[1] / 2;
    const long long* src = ei;
    const long long* dst = ei + E;

    // workspace carve-up (256B aligned)
    char* ws = (char*)d_ws;
    size_t off = 0;
    auto carve = [&](size_t bytes) -> char* {
        char* p = ws + off;
        off = (off + bytes + 255) & ~(size_t)255;
        return p;
    };
    float*          dinv = (float*)         carve((size_t)NNODES * 4);
    unsigned short* xb   = (unsigned short*)carve((size_t)NNODES * FEAT * 2);
    float*          h    = (float*)         carve((size_t)NNODES * FEAT * 4);
    float*          agg  = (float*)         carve((size_t)NNODES * FEAT * 4);
    unsigned short* rb   = (unsigned short*)carve((size_t)NNODES * FEAT * 2);
    unsigned short* Wp1  = (unsigned short*)carve((size_t)FEAT * FEAT * 2);
    unsigned short* Wp2  = (unsigned short*)carve((size_t)FEAT * FEAT * 2);

    const int total = NNODES * FEAT;            // 6,400,000
    const dim3 blk(256);
    const int gN    = (NNODES + 255) / 256;
    const int gE    = (E + 255) / 256;
    const int gTot  = (total + 255) / 256;
    const int gEdge = (E * 32 + 255) / 256;     // one wave per edge
    const int gGemm = NNODES / 16;              // 3125, exact
    const int gPack = (FEAT * FEAT) / 256;      // 64

    // --- normalization ---
    deg_init_kernel<<<gN, blk, 0, stream>>>(dinv, NNODES);
    deg_accum_kernel<<<gE, blk, 0, stream>>>(dst, dinv, E);
    deg_rsqrt_kernel<<<gN, blk, 0, stream>>>(dinv, NNODES);

    // --- precision prep ---
    cvt_f32_bf16_kernel<<<gTot, blk, 0, stream>>>(x, xb, total);
    pack_w_kernel<<<gPack, blk, 0, stream>>>(W1, Wp1);
    pack_w_kernel<<<gPack, blk, 0, stream>>>(W2, Wp2);

    // --- layer 1: h = x@W1 ; agg = A_norm h ; relu(agg+b1) -> bf16 ---
    gemm_bf16_wmma_kernel<<<gGemm, blk, 0, stream>>>(xb, Wp1, h);
    self_init_kernel<<<gTot, blk, 0, stream>>>(h, dinv, agg, total);
    edge_scatter_kernel<<<gEdge, blk, 0, stream>>>(src, dst, dinv, h, agg, E);
    bias_relu_bf16_kernel<<<gTot, blk, 0, stream>>>(agg, b1, rb, total);

    // --- layer 2: h = relu1@W2 ; out = A_norm h + b2 ---
    gemm_bf16_wmma_kernel<<<gGemm, blk, 0, stream>>>(rb, Wp2, h);
    self_init_kernel<<<gTot, blk, 0, stream>>>(h, dinv, out, total);
    edge_scatter_kernel<<<gEdge, blk, 0, stream>>>(src, dst, dinv, h, out, E);
    bias_add_kernel<<<gTot, blk, 0, stream>>>(out, b2, total);
}
